// MultiScaleTemporalSpatioTemporalGNN_40037685133360
// MI455X (gfx1250) — matrix operand order repaired and checked by reading, not compiled, not run on hardware
//
#include <hip/hip_runtime.h>
#include <hip/hip_bf16.h>
#include <math.h>

typedef _Float16 v16h __attribute__((ext_vector_type(16)));
typedef _Float16 v8h  __attribute__((ext_vector_type(8)));
typedef _Float16 v4h  __attribute__((ext_vector_type(4)));
typedef float    v8f  __attribute__((ext_vector_type(8)));

#define HD   256   // hidden dim H
#define TT   6     // T (B=1)
#define FIN  11
#define SHD  64    // per-scale LSTM hidden

__device__ __forceinline__ float sigmoidf_(float x){ return 1.0f/(1.0f+__expf(-x)); }

// ============================================================================
// Weight prep: WT16[n*K + k] = (f16) W[k*N + n]   (transpose + convert, once)
// ============================================================================
__global__ void k_wcvt(const float* __restrict__ W, _Float16* __restrict__ WT,
                       int K, int N){
  long idx = (long)blockIdx.x*blockDim.x + threadIdx.x;
  if (idx >= (long)K*N) return;
  int k = (int)(idx / N), n = (int)(idx - (long)k*N);
  WT[(long)n*K + k] = (_Float16)W[idx];
}

// ============================================================================
// Generic WMMA GEMM: C[M,N] = act(A[M,K1+K2] @ W + bias + eattr@wedge)
//  - A fp32, split into A1 (first K1 cols) / A2 (next K2); optional row-gather
//  - WT fp16 column-major [N, K] (ldw = K), pre-converted
//  - optional scatter: atomicAdd rows into C[scat[row]] (fused segment_sum)
//  Tile: 128x64 per 256-thread block; 8 waves 4x2; wave 32x32 via 2x2 WMMA.
//  Double-buffered LDS; A rows past M are clamped to row M-1 (their products only
//  reach output rows >= M, which the epilogue never stores) -> branch-free.
// ============================================================================
template<int ACT>   // 0=none 1=relu 2=sigmoid
__global__ __launch_bounds__(256)
void k_gemm(const float* __restrict__ A1, long lda1, const int* __restrict__ g1,
            const float* __restrict__ A2, long lda2, const int* __restrict__ g2,
            int K1, int K2,
            const _Float16* __restrict__ WT, long ldw,
            const float* __restrict__ bias,
            const float* __restrict__ eattr, const float* __restrict__ wedge,
            float* __restrict__ C, long ldc, const int* __restrict__ scat,
            int M, int N)
{
  __shared__ __align__(16) _Float16 As[2][128*40];
  __shared__ __align__(16) _Float16 Bs[2][64*40];
  const int tid  = threadIdx.x;
  const int lane = tid & 31;
  const int w    = tid >> 5;
  const int wm   = w >> 1;      // 0..3 (M dir)
  const int wn   = w & 1;       // 0..1 (N dir)
  const int lr   = lane & 15;
  const int lh   = lane >> 4;   // 0/1 lane half

  const int m0 = blockIdx.y * 128;
  const int n0 = blockIdx.x * 64;
  const int K  = K1 + K2;

  // ---- hoist all A-row addressing out of the K loop (4 load slots) ----
  const float* aptr1[4];
  const float* aptr2[4];
  int arowlds[4];
  #pragma unroll
  for (int i=0;i<4;i++){
    int q = tid + i*256;
    int row = q >> 3, seg = q & 7;
    arowlds[i] = row*40 + seg*4;
    int gr = m0 + row;
    int grc = (gr < M) ? gr : (M-1);      // clamp: safe, only feeds dead rows
    long r1 = g1 ? (long)g1[grc] : (long)grc;
    aptr1[i] = A1 + r1*lda1 + seg*4;
    if (K2 > 0) {
      long r2 = g2 ? (long)g2[grc] : (long)grc;
      aptr2[i] = A2 + r2*lda2 + seg*4;
    } else {
      aptr2[i] = aptr1[i];
    }
  }
  // B staging slot: one v8h per thread per K-step (64 cols x 4 chunks)
  const int bcol0 = tid >> 2;
  const int bk8   = (tid & 3) * 8;
  const _Float16* wbase = WT + (long)(n0 + bcol0)*ldw + bk8;
  const int blds = bcol0*40 + bk8;

  auto stage = [&](int kb, int buf){
    const bool first = (kb < K1);
    const int  kloc  = first ? kb : (kb - K1);
    #pragma unroll
    for (int i=0;i<4;i++){
      const float* p = (first ? aptr1[i] : aptr2[i]) + kloc;
      const float4 f = *(const float4*)p;
      v4h hv;
      hv[0]=(_Float16)f.x; hv[1]=(_Float16)f.y; hv[2]=(_Float16)f.z; hv[3]=(_Float16)f.w;
      *(v4h*)&As[buf][arowlds[i]] = hv;
    }
    v8h bv = *(const v8h*)(wbase + kb);
    *(v8h*)&Bs[buf][blds] = bv;
  };

  v8f acc[2][2];
  #pragma unroll
  for (int i=0;i<2;i++)
    #pragma unroll
    for (int j=0;j<2;j++)
      #pragma unroll
      for (int e=0;e<8;e++) acc[i][j][e] = 0.0f;

  stage(0, 0);
  __syncthreads();

  for (int kb = 0; kb < K; kb += 32) {
    const int cur = (kb >> 5) & 1;
    if (kb + 32 < K) stage(kb + 32, cur ^ 1);   // prefetch next tile
    // ---- fragments per CDNA5 16x16x32 f16 layout + WMMA ----
    #pragma unroll
    for (int i=0;i<2;i++){
      int arow = wm*32 + i*16 + lr;
      v8h alo = *(const v8h*)&As[cur][arow*40 + lh*8];        // K 0..7 / 8..15
      v8h ahi = *(const v8h*)&As[cur][arow*40 + lh*8 + 16];   // K 16..23 / 24..31
      v16h a;
      #pragma unroll
      for (int e=0;e<8;e++){ a[e]=alo[e]; a[e+8]=ahi[e]; }
      #pragma unroll
      for (int j=0;j<2;j++){
        int bcol = wn*32 + j*16 + lr;
        v8h blo = *(const v8h*)&Bs[cur][bcol*40 + lh*8];      // K 0..7 / 8..15
        v8h bhi = *(const v8h*)&Bs[cur][bcol*40 + lh*8 + 16]; // K 16..23 / 24..31
        v16h b;
        #pragma unroll
        for (int e=0;e<8;e++){ b[e]=blo[e]; b[e+8]=bhi[e]; }
        acc[i][j] = __builtin_amdgcn_wmma_f32_16x16x32_f16(
            false, a, false, b, (short)0, acc[i][j], false, false);
      }
    }
    __syncthreads();
  }

  // ---- epilogue: bias, edge-attr K=3 part, activation, store/scatter ----
  #pragma unroll
  for (int i=0;i<2;i++){
    #pragma unroll
    for (int j=0;j<2;j++){
      int colg = n0 + wn*32 + j*16 + lr;
      if (colg >= N) continue;
      #pragma unroll
      for (int r=0;r<8;r++){
        int rowg = m0 + wm*32 + i*16 + lh*8 + r;
        if (rowg >= M) continue;
        float val = acc[i][j][r];
        if (bias) val += bias[colg];
        if (eattr) {
          const float* e3 = eattr + (long)rowg*3;
          val += e3[0]*wedge[colg] + e3[1]*wedge[N+colg] + e3[2]*wedge[2*N+colg];
        }
        if (ACT==1)      val = fmaxf(val, 0.0f);
        else if (ACT==2) val = sigmoidf_(val);
        if (scat) atomicAdd(&C[(long)scat[rowg]*ldc + colg], val);
        else      C[(long)rowg*ldc + colg] = val;
      }
    }
  }
}

// ============================== small kernels ===============================
__global__ void k_zero(float* p, long n){
  long i = (long)blockIdx.x*blockDim.x + threadIdx.x;
  if (i < n) p[i] = 0.f;
}

__global__ void k_encoder(const float* __restrict__ x, const float* __restrict__ w,
                          const float* __restrict__ b, float* __restrict__ h, long M){
  long idx = (long)blockIdx.x*blockDim.x + threadIdx.x;
  if (idx >= M*HD) return;
  long row = idx / HD; int c = (int)(idx - row*HD);
  float acc = b[c];
  #pragma unroll
  for (int f=0; f<FIN; f++) acc += x[row*FIN+f]*w[f*HD+c];
  h[idx] = fmaxf(acc, 0.f);
}

__global__ void k_combine(const float* __restrict__ gate, float* __restrict__ upd,
                          const float* __restrict__ h, long n){
  long i = (long)blockIdx.x*blockDim.x + threadIdx.x;
  if (i < n){ float g = gate[i]; upd[i] = g*upd[i] + (1.f-g)*h[i]; }
}

// out[row*ldo+offo+c] = LN( xa[row*lda+offa+c] + xb?[row*ldb+offb+c] ) * g + b
__global__ void k_resid_ln(const float* __restrict__ xa, long lda, long offa,
                           const float* __restrict__ xb, long ldb, long offb,
                           const float* __restrict__ g, const float* __restrict__ bt,
                           float* __restrict__ out, long ldo, long offo, int D)
{
  __shared__ float s1[256], s2[256];
  int c = threadIdx.x; long row = blockIdx.x;
  float v = 0.f;
  if (c < D) {
    v = xa[row*lda + offa + c];
    if (xb) v += xb[row*ldb + offb + c];
  }
  s1[c] = v; s2[c] = v*v;
  __syncthreads();
  for (int s=128; s>0; s>>=1){
    if (c < s){ s1[c] += s1[c+s]; s2[c] += s2[c+s]; }
    __syncthreads();
  }
  float mean = s1[0]/D;
  float var  = s2[0]/D - mean*mean;
  float inv  = rsqrtf(var + 1e-5f);
  if (c < D) out[row*ldo + offo + c] = (v-mean)*inv*g[c] + bt[c];
}

// h [T,N,H] -> hseq [N,T,H]
__global__ void k_transpose(const float* __restrict__ h, float* __restrict__ hseq, int Nn){
  long idx = (long)blockIdx.x*blockDim.x + threadIdx.x;
  long total = (long)TT*Nn*HD;
  if (idx >= total) return;
  long n = idx/((long)TT*HD); long rem = idx - n*TT*HD;
  int t = (int)(rem/HD); int c = (int)(rem - (long)t*HD);
  hseq[idx] = h[((long)t*Nn + n)*HD + c];
}

// temporal mean-pool with repeat-last padding: vseq [N,Tp,H]
__global__ void k_pool(const float* __restrict__ hseq, float* __restrict__ vseq,
                       int sc, int Tp, int Nn){
  long idx = (long)blockIdx.x*blockDim.x + threadIdx.x;
  long total = (long)Nn*Tp*HD;
  if (idx >= total) return;
  long n = idx/((long)Tp*HD); long rem = idx - n*Tp*HD;
  int tp = (int)(rem/HD); int c = (int)(rem - (long)tp*HD);
  float s = 0.f;
  for (int k=0; k<sc; k++){
    int t = tp*sc + k; if (t > TT-1) t = TT-1;
    s += hseq[((long)n*TT + t)*HD + c];
  }
  vseq[idx] = s / (float)sc;
}

// LSTM cell: gates = gbuf (h@wh) + xproj_t (v@wi) + bi + bh; order i,f,g,o
__global__ void k_lstm_cell(const float* __restrict__ gbuf, const float* __restrict__ xproj,
                            const float* __restrict__ bi, const float* __restrict__ bh,
                            float* __restrict__ hstate, float* __restrict__ cstate,
                            float* __restrict__ outseq, int Nn, int Tp, int t){
  long idx = (long)blockIdx.x*blockDim.x + threadIdx.x;
  if (idx >= (long)Nn*SHD) return;
  long n = idx / SHD; int j = (int)(idx - n*SHD);
  const float* gr = gbuf  + n*(long)(4*SHD);
  const float* xp = xproj + ((long)n*Tp + t)*(4*SHD);
  float gi = gr[j]         + xp[j]         + bi[j]         + bh[j];
  float gf = gr[SHD+j]     + xp[SHD+j]     + bi[SHD+j]     + bh[SHD+j];
  float gg = gr[2*SHD+j]   + xp[2*SHD+j]   + bi[2*SHD+j]   + bh[2*SHD+j];
  float go = gr[3*SHD+j]   + xp[3*SHD+j]   + bi[3*SHD+j]   + bh[3*SHD+j];
  float c  = sigmoidf_(gf)*cstate[idx] + sigmoidf_(gi)*tanhf(gg);
  float hh = sigmoidf_(go)*tanhf(c);
  cstate[idx] = c; hstate[idx] = hh;
  outseq[((long)n*Tp + t)*SHD + j] = hh;
}

// per-scale MHA, last query only. 4 heads x hd=16, seq Tp<=6
__global__ void k_attn(const float* __restrict__ qbuf, const float* __restrict__ kbuf,
                       const float* __restrict__ vbuf, float* __restrict__ ctx,
                       int Nn, int Tp){
  long idx = (long)blockIdx.x*blockDim.x + threadIdx.x;
  if (idx >= (long)Nn*4) return;
  long n = idx >> 2; int off = (int)(idx & 3) * 16;
  const float* q = qbuf + n*64 + off;
  float s[6]; float mx = -1e30f;
  for (int t=0; t<Tp; t++){
    const float* kk = kbuf + ((long)n*Tp + t)*64 + off;
    float d = 0.f;
    #pragma unroll
    for (int e=0;e<16;e++) d += q[e]*kk[e];
    d *= 0.25f;                       // 1/sqrt(16)
    s[t] = d; mx = fmaxf(mx, d);
  }
  float den = 0.f;
  for (int t=0; t<Tp; t++){ s[t] = __expf(s[t]-mx); den += s[t]; }
  float inv = 1.f/den;
  #pragma unroll
  for (int e=0;e<16;e++){
    float a = 0.f;
    for (int t=0; t<Tp; t++) a += s[t]*vbuf[((long)n*Tp + t)*64 + off + e];
    ctx[n*64 + off + e] = a*inv;
  }
}

// head second GEMM: N=3 columns
__global__ void k_head2(const float* __restrict__ h1, const float* __restrict__ w2,
                        const float* __restrict__ b2, float* __restrict__ out, int Nn){
  long idx = (long)blockIdx.x*blockDim.x + threadIdx.x;
  if (idx >= (long)Nn*3) return;
  long n = idx/3; int j = (int)(idx - n*3);
  float acc = b2[j];
  for (int k=0; k<HD; k++) acc += h1[n*HD + k]*w2[k*3 + j];
  out[idx] = acc;
}

// ============================== host orchestration ==========================
extern "C" void kernel_launch(void* const* d_in, const int* in_sizes, int n_in,
                              void* d_out, int out_size, void* d_ws, size_t ws_size,
                              hipStream_t stream) {
  const float* x     = (const float*)d_in[0];
  const float* eattr = (const float*)d_in[1];
  const int Nn = in_sizes[0] / (TT*FIN);   // 20000
  const int E  = in_sizes[1] / 3;          // 80000

  // params flattened in jax pytree order (sorted dict keys):
  // [2]=enc_b [3]=enc_w  [4..13]=fusion{bk,bo,bq,bv,ln_b,ln_g,wk,wo,wq,wv}
  // [14..17]=head{b1,b2,w1,w2}  [18..41]=lstm[3][2](wi,wh,bi,bh)
  // [42..113]=mp[6]{g_b,g_w,ln_b,ln_g,m_b1,m_b2,m_w1,m_w2,u_b1,u_b2,u_w1,u_w2}
  // [114..143]=t_attn[3]{bk,bo,bq,bv,ln_b,ln_g,wk,wo,wq,wv}  [144]=edge_index
  auto pf = [&](int i)->const float* { return (const float*)d_in[i]; };
  const int FUSB = 4, HEADB = 14, LSTMB = 18, MPB = 42, TATB = 114;
  const int* ei  = (const int*)d_in[144];
  const int* src = ei;
  const int* dst = ei + E;

  // workspace bump allocator
  char* wsp = (char*)d_ws;
  auto alloc = [&](size_t bytes)->void* {
    char* p = wsp; wsp += (bytes + 255) & ~(size_t)255; return (void*)p;
  };
  const size_t slab = (size_t)TT*Nn*HD*sizeof(float);
  float* h    = (float*)alloc(slab);
  float* agg  = (float*)alloc(slab);
  float* buf1 = (float*)alloc(slab);
  float* buf2 = (float*)alloc(slab);
  float* buf3 = (float*)alloc(slab);
  float* ebuf = (float*)alloc((size_t)E*HD*sizeof(float));
  // phase-2 aliases (free after message passing)
  float* hseq  = agg;
  float* vseq  = buf1;
  float* xproj = buf2;
  float* out0  = buf3;
  float* out1  = ebuf;   // Nn*6*64 floats <= E*HD floats
  // small persistent buffers
  float* gbuf   = (float*)alloc((size_t)Nn*4*SHD*sizeof(float));
  float* hstate = (float*)alloc((size_t)Nn*SHD*sizeof(float));
  float* cstate = (float*)alloc((size_t)Nn*SHD*sizeof(float));
  float* kbuf   = (float*)alloc((size_t)Nn*TT*SHD*sizeof(float));
  float* vbuf   = (float*)alloc((size_t)Nn*TT*SHD*sizeof(float));
  float* qbuf   = (float*)alloc((size_t)Nn*SHD*sizeof(float));
  float* ctx    = (float*)alloc((size_t)Nn*SHD*sizeof(float));
  float* abuf   = (float*)alloc((size_t)Nn*SHD*sizeof(float));
  float* fin    = (float*)alloc((size_t)Nn*192*sizeof(float));
  float* fv     = (float*)alloc((size_t)Nn*192*sizeof(float));
  float* fo     = (float*)alloc((size_t)Nn*192*sizeof(float));
  float* fusedb = (float*)alloc((size_t)Nn*192*sizeof(float));
  float* h1b    = (float*)alloc((size_t)Nn*HD*sizeof(float));

  auto zero = [&](float* p, long n){
    k_zero<<<(unsigned)((n+255)/256), 256, 0, stream>>>(p, n);
  };
  // one-time fp32 -> fp16 col-major weight conversion
  auto prep = [&](const float* W, int K, int N)->const _Float16* {
    _Float16* WT = (_Float16*)alloc((size_t)K*N*sizeof(_Float16));
    long total = (long)K*N;
    k_wcvt<<<(unsigned)((total+255)/256), 256, 0, stream>>>(W, WT, K, N);
    return WT;
  };
  auto gemm = [&](int act, const float* A1, long lda1, const int* g1,
                  const float* A2, long lda2, const int* g2,
                  int K1, int K2, const _Float16* WT, const float* bias,
                  const float* ea, const float* we,
                  float* C, long ldc, const int* scat, int M, int Nw){
    long ldw = K1 + K2;
    dim3 grid((Nw+63)/64, (M+127)/128);
    if (act==0)      k_gemm<0><<<grid,256,0,stream>>>(A1,lda1,g1,A2,lda2,g2,K1,K2,WT,ldw,bias,ea,we,C,ldc,scat,M,Nw);
    else if (act==1) k_gemm<1><<<grid,256,0,stream>>>(A1,lda1,g1,A2,lda2,g2,K1,K2,WT,ldw,bias,ea,we,C,ldc,scat,M,Nw);
    else             k_gemm<2><<<grid,256,0,stream>>>(A1,lda1,g1,A2,lda2,g2,K1,K2,WT,ldw,bias,ea,we,C,ldc,scat,M,Nw);
  };

  // ---------------- weight prep (f16, transposed) ----------------
  const _Float16 *W_mw1[6], *W_mw2[6], *W_gw[6], *W_uw1[6], *W_uw2[6];
  for (int L=0; L<6; L++){
    int pb = MPB + 12*L;
    W_mw1[L] = prep(pf(pb+6),  2*HD, HD);   // first 512 rows of m_w1
    W_mw2[L] = prep(pf(pb+7),  HD,   HD);
    W_gw [L] = prep(pf(pb+1),  2*HD, HD);
    W_uw1[L] = prep(pf(pb+10), 2*HD, HD);
    W_uw2[L] = prep(pf(pb+11), HD,   HD);
  }
  const _Float16 *W_wi[3][2], *W_wh[3][2];
  for (int si=0; si<3; si++)
    for (int l=0; l<2; l++){
      int lb = LSTMB + si*8 + l*4;
      W_wi[si][l] = prep(pf(lb+0), (l==0)?HD:SHD, 4*SHD);
      W_wh[si][l] = prep(pf(lb+1), SHD, 4*SHD);
    }
  const _Float16 *W_ak[3], *W_av[3], *W_aq[3], *W_ao[3];
  for (int si=0; si<3; si++){
    int tb = TATB + 10*si;
    W_ak[si] = prep(pf(tb+6), SHD, SHD);
    W_ao[si] = prep(pf(tb+7), SHD, SHD);
    W_aq[si] = prep(pf(tb+8), SHD, SHD);
    W_av[si] = prep(pf(tb+9), SHD, SHD);
  }
  const _Float16* W_fwv = prep(pf(FUSB+9), 192, 192);
  const _Float16* W_fwo = prep(pf(FUSB+7), 192, 192);
  const _Float16* W_hw1 = prep(pf(HEADB+2), 192, HD);

  // ---------------- encoder ----------------
  {
    long M = (long)TT*Nn;
    k_encoder<<<(unsigned)((M*HD+255)/256), 256, 0, stream>>>(x, pf(3), pf(2), h, M);
  }

  // ---------------- 6 message-passing layers ----------------
  for (int L=0; L<6; L++){
    int pb = MPB + 12*L;
    const float *g_b=pf(pb+0), *ln_b=pf(pb+2), *ln_g=pf(pb+3),
                *m_b1=pf(pb+4), *m_b2=pf(pb+5), *m_w1f=pf(pb+6),
                *u_b1=pf(pb+8), *u_b2=pf(pb+9);
    zero(agg, (long)TT*Nn*HD);
    for (int g=0; g<TT; g++){
      const float* hg = h + (size_t)g*Nn*HD;
      // pre = relu(h[src]@W1a + h[dst]@W1b + e@W1e + b1)
      gemm(1, hg, HD, src, hg, HD, dst, HD, HD, W_mw1[L], m_b1,
           eattr, m_w1f + (size_t)2*HD*HD, ebuf, HD, nullptr, E, HD);
      // agg[dst] += pre@m_w2 + b2   (fused segment_sum via atomics)
      gemm(0, ebuf, HD, nullptr, nullptr, 0, nullptr, HD, 0, W_mw2[L], m_b2,
           nullptr, nullptr, agg + (size_t)g*Nn*HD, HD, dst, E, HD);
    }
    int M = TT*Nn;
    gemm(2, h, HD, nullptr, agg, HD, nullptr, HD, HD, W_gw[L],  g_b,  nullptr,nullptr, buf1, HD, nullptr, M, HD);
    gemm(1, h, HD, nullptr, agg, HD, nullptr, HD, HD, W_uw1[L], u_b1, nullptr,nullptr, buf2, HD, nullptr, M, HD);
    gemm(0, buf2, HD, nullptr, nullptr,0,nullptr, HD, 0,   W_uw2[L], u_b2, nullptr,nullptr, buf3, HD, nullptr, M, HD);
    k_combine<<<(unsigned)(((long)M*HD+255)/256), 256, 0, stream>>>(buf1, buf3, h, (long)M*HD);
    k_resid_ln<<<(unsigned)M, 256, 0, stream>>>(buf3, HD, 0, nullptr,0,0, ln_g, ln_b, h, HD, 0, HD);
  }

  // ---------------- reorder to per-node sequences ----------------
  {
    long total = (long)TT*Nn*HD;
    k_transpose<<<(unsigned)((total+255)/256), 256, 0, stream>>>(h, hseq, Nn);
  }

  // ---------------- per-scale LSTM + attention ----------------
  const int scales[3] = {1, 2, 4};
  for (int si=0; si<3; si++){
    int sc = scales[si];
    int Tp = (TT + sc - 1)/sc;   // 6,3,2
    {
      long total = (long)Nn*Tp*HD;
      k_pool<<<(unsigned)((total+255)/256), 256, 0, stream>>>(hseq, vseq, sc, Tp, Nn);
    }
    for (int layer=0; layer<2; layer++){
      int lb = LSTMB + si*8 + layer*4;
      const float *bi=pf(lb+2), *bh=pf(lb+3);
      int din = (layer==0) ? HD : SHD;
      const float* inseq = (layer==0) ? vseq : out0;
      float* oseq = (layer==0) ? out0 : out1;
      // xproj = inseq @ wi  [Nn*Tp, 4*SHD]
      gemm(0, inseq, din, nullptr, nullptr,0,nullptr, din, 0, W_wi[si][layer], nullptr,
           nullptr,nullptr, xproj, 4*SHD, nullptr, Nn*Tp, 4*SHD);
      zero(hstate, (long)Nn*SHD);
      zero(cstate, (long)Nn*SHD);
      for (int t=0; t<Tp; t++){
        gemm(0, hstate, SHD, nullptr, nullptr,0,nullptr, SHD, 0, W_wh[si][layer], nullptr,
             nullptr,nullptr, gbuf, 4*SHD, nullptr, Nn, 4*SHD);
        k_lstm_cell<<<(unsigned)(((long)Nn*SHD+255)/256), 256, 0, stream>>>(
            gbuf, xproj, bi, bh, hstate, cstate, oseq, Nn, Tp, t);
      }
    }
    // attention over out1 [Nn,Tp,64], only last query needed
    int tb = TATB + 10*si;
    const float *bk=pf(tb+0), *bo=pf(tb+1), *bq=pf(tb+2), *bv=pf(tb+3),
                *alnb=pf(tb+4), *alng=pf(tb+5);
    gemm(0, out1, SHD, nullptr, nullptr,0,nullptr, SHD, 0, W_ak[si], bk, nullptr,nullptr,
         kbuf, SHD, nullptr, Nn*Tp, SHD);
    gemm(0, out1, SHD, nullptr, nullptr,0,nullptr, SHD, 0, W_av[si], bv, nullptr,nullptr,
         vbuf, SHD, nullptr, Nn*Tp, SHD);
    gemm(0, out1 + (size_t)(Tp-1)*SHD, (long)Tp*SHD, nullptr, nullptr,0,nullptr, SHD, 0,
         W_aq[si], bq, nullptr,nullptr, qbuf, SHD, nullptr, Nn, SHD);
    k_attn<<<(unsigned)(((long)Nn*4+127)/128), 128, 0, stream>>>(qbuf, kbuf, vbuf, ctx, Nn, Tp);
    gemm(0, ctx, SHD, nullptr, nullptr,0,nullptr, SHD, 0, W_ao[si], bo, nullptr,nullptr,
         abuf, SHD, nullptr, Nn, SHD);
    // fin[:, si*64:(si+1)*64] = LN(o_last + attn_out)
    k_resid_ln<<<(unsigned)Nn, 256, 0, stream>>>(
        out1, (long)Tp*SHD, (long)(Tp-1)*SHD, abuf, SHD, 0,
        alng, alnb, fin, 192, (long)si*SHD, SHD);
  }

  // ---------------- fusion attention (S=1 => softmax == 1) ----------------
  {
    const float *fbo=pf(FUSB+1), *fbv=pf(FUSB+3), *flnb=pf(FUSB+4), *flng=pf(FUSB+5);
    gemm(0, fin, 192, nullptr, nullptr,0,nullptr, 192, 0, W_fwv, fbv, nullptr,nullptr,
         fv, 192, nullptr, Nn, 192);
    gemm(0, fv, 192, nullptr, nullptr,0,nullptr, 192, 0, W_fwo, fbo, nullptr,nullptr,
         fo, 192, nullptr, Nn, 192);
    k_resid_ln<<<(unsigned)Nn, 256, 0, stream>>>(fin, 192, 0, fo, 192, 0,
                                                 flng, flnb, fusedb, 192, 0, 192);
  }

  // ---------------- head ----------------
  {
    const float *hb1=pf(HEADB+0), *hb2=pf(HEADB+1), *hw2=pf(HEADB+3);
    gemm(1, fusedb, 192, nullptr, nullptr,0,nullptr, 192, 0, W_hw1, hb1, nullptr,nullptr,
         h1b, HD, nullptr, Nn, HD);
    k_head2<<<(unsigned)(((long)Nn*3+255)/256), 256, 0, stream>>>(h1b, hw2, hb2, (float*)d_out, Nn);
  }
}
